// SetCriterion_84997402788702
// MI455X (gfx1250) — matrix operand order repaired and checked by reading, not compiled, not run on hardware
//
#include <hip/hip_runtime.h>

#ifndef __has_builtin
#define __has_builtin(x) 0
#endif

typedef __attribute__((ext_vector_type(16))) _Float16 v16h;
typedef __attribute__((ext_vector_type(8)))  float    v8f;
typedef __attribute__((ext_vector_type(4)))  unsigned int v4u;
typedef __attribute__((ext_vector_type(8)))  int      v8i;
typedef __attribute__((ext_vector_type(4)))  int      v4i;

namespace {
constexpr int L = 10, Bn = 2, Qn = 100, P = 16384, Nn = 20, Cc = 134;
constexpr int QT = 7;                       // ceil(Q/16) q-tiles of 16 rows
constexpr int WAVES = 16;                   // waves per block in k_pairwise
constexpr int BLK   = WAVES * 32;           // 512 threads
constexpr int CHUNK = P / WAVES;            // 1024 pixels per wave
constexpr float SMOOTH = 1.0f;
constexpr float MASK_SCALE = 5.0f, DICE_SCALE = 5.0f, CE_SCALE = 2.0f;
constexpr float INFC = 1e30f;
constexpr float LOG2E = 1.44269504088896f;
constexpr float LN2   = 0.69314718055995f;
}

// ---------------------------------------------------------------------------
// K0: per-batch pixel counts of each GT segment (G.sum over pixels).
// ---------------------------------------------------------------------------
__global__ __launch_bounds__(256) void k_gsum(const int* __restrict__ seg,
                                              float* __restrict__ gsum) {
  __shared__ int h[32];
  const int t = threadIdx.x;
  if (t < 32) h[t] = 0;
  __syncthreads();
  const int b = blockIdx.x;
  for (int p = t; p < P; p += 256) {
    int s = seg[b * P + p];
    if (s >= 1 && s <= Nn) atomicAdd(&h[s - 1], 1);
  }
  __syncthreads();
  if (t < 32) gsum[b * 32 + t] = (float)h[t];
}

// ---------------------------------------------------------------------------
// K1: heavy kernel. One block (16 waves) per (l,b,q-tile). The pan_seg row
// (64 KB) is staged into LDS once per block via the Tensor Data Mover
// (tensor_load_to_lds + s_wait_tensorcnt); each wave then streams 1024
// pixels, computes softplus(-x)/softplus(x)/sigmoid(x) as f16 A-tiles and
// reduces them against a one-hot B-tile (cols 0..19 = segments, logical col
// 20 = all-ones -> row totals) with v_wmma_f32_16x16x32_f16. The seg buffer
// and the 96 KB reduction arena share LDS (disjoint lifetimes).
// ---------------------------------------------------------------------------
__global__ __launch_bounds__(BLK) void k_pairwise(
    const float* __restrict__ x, const int* __restrict__ seg,
    const float* __restrict__ gsum, float* __restrict__ bce,
    float* __restrict__ dice)
{
  __shared__ union {
    int   seg[P];                 // 64 KB, live during the main loop
    float red[6][WAVES][256];     // 96 KB, live after the post-loop barrier
  } sm;

  const int wg   = blockIdx.x;             // 0 .. L*B*QT-1
  const int qt   = wg % QT;
  const int lb   = wg / QT;                // l*B + b
  const int b    = lb % Bn;
  const int t    = threadIdx.x;
  const int wv   = t >> 5;                 // wave id 0..15
  const int lane = t & 31;
  const int mrow = lane & 15;              // A row / B column
  const int half = lane >> 4;
  const int kbase = (lane >> 4) * 8;       // B-fragment K base
  const int q0   = qt * 16;

  // ---- stage pan_seg[b] row into LDS (TDM async-tensor path) --------------
#if defined(__gfx1250__) && __has_builtin(__builtin_amdgcn_tensor_load_to_lds)
  if (wv == 0) {
    const unsigned lds_off = (unsigned)(uintptr_t)(void*)sm.seg;  // flat lo32 = LDS offset
    const unsigned long long ga =
        (unsigned long long)(uintptr_t)(seg + (size_t)b * P);
    v4u g0;
    g0.x = 1u;                                            // count=1, user D#
    g0.y = lds_off;                                       // lds_addr (bytes)
    g0.z = (unsigned)ga;                                  // global_addr[31:0]
    g0.w = (unsigned)((ga >> 32) & 0x01FFFFFFu) | (2u << 30);  // addr[56:32]|type=2
    v8i g1;
    g1[0] = (int)(2u << 16);                              // data_size = 4 bytes
    g1[1] = (int)((unsigned)(P & 0xFFFF) << 16);          // tensor_dim0 lo16
    g1[2] = (int)((unsigned)(P >> 16) | (1u << 16));      // dim0 hi | tensor_dim1=1
    g1[3] = (int)((unsigned)(P & 0xFFFF) << 16);          // dim1 hi=0 | tile_dim0=P
    g1[4] = 1;                                            // tile_dim1=1, tile_dim2=0
    g1[5] = P;                                            // tensor_dim0_stride lo32
    g1[6] = 0;
    g1[7] = 0;
    v4i g2 = {}; v4i g3 = {};
#if defined(__clang_major__) && (__clang_major__ >= 23)
    v8i gx = {};
    __builtin_amdgcn_tensor_load_to_lds(g0, g1, g2, g3, gx, 0);
#else
    __builtin_amdgcn_tensor_load_to_lds(g0, g1, g2, g3, 0);
#endif
    __builtin_amdgcn_s_wait_tensorcnt(0);
  }
#else
  for (int i = t; i < P / 4; i += BLK)
    ((int4*)sm.seg)[i] = ((const int4*)(seg + (size_t)b * P))[i];
#endif
  __syncthreads();

  int qload = q0 + mrow;                   // clamp padded rows (discarded later)
  if (qload > Qn - 1) qload = Qn - 1;
  const float* rowp = x + ((size_t)lb * Qn + qload) * (size_t)P + half * 8;
  const int*   sseg = sm.seg;

  v8f aP0 = {}, aP1 = {}, aN0 = {}, aN1 = {}, aI0 = {}, aI1 = {};
  const int  n0    = mrow;                 // tile0 logical column 0..15
  const int  n1    = 16 + mrow;            // tile1 logical column 16..31
  const bool ones1 = (n1 == 20);           // all-ones "row total" column

  // scalarize trip count (wave-uniform) -> plain scalar loop, no EXEC masking
  const int pbeg = __builtin_amdgcn_readfirstlane(wv * CHUNK);
  const int pend = pbeg + CHUNK;

  for (int p0 = pbeg; p0 < pend; p0 += 32) {
    __builtin_prefetch(rowp + p0 + 512, 0, 0);   // global_prefetch_b8

    // A tile (16-bit 16x32 layout): lanes 0-15 K={0..7,16..23},
    // lanes 16-31 K={8..15,24..31} (via rowp's +half*8).
    const float4 f0 = *reinterpret_cast<const float4*>(rowp + p0);
    const float4 f1 = *reinterpret_cast<const float4*>(rowp + p0 + 4);
    const float4 f2 = *reinterpret_cast<const float4*>(rowp + p0 + 16);
    const float4 f3 = *reinterpret_cast<const float4*>(rowp + p0 + 20);
    float xv[16] = {f0.x, f0.y, f0.z, f0.w, f1.x, f1.y, f1.z, f1.w,
                    f2.x, f2.y, f2.z, f2.w, f3.x, f3.y, f3.z, f3.w};

    v16h Ap, An, As;
#pragma unroll
    for (int e = 0; e < 16; ++e) {
      float xx  = xv[e];
      float ee  = __builtin_amdgcn_exp2f(__builtin_fabsf(xx) * -LOG2E);
      float ope = 1.0f + ee;                              // in (1,2]
      float lp  = __builtin_amdgcn_logf(ope) * LN2;       // log1p(exp(-|x|))
      float sp_pos = fmaxf(-xx, 0.0f) + lp;               // softplus(-x)
      float sp_neg = sp_pos + xx;                         // softplus(x)
      float rr  = __builtin_amdgcn_rcpf(ope);
      float sg  = (xx >= 0.0f) ? rr : (1.0f - rr);        // sigmoid(x)
      Ap[e] = (_Float16)sp_pos;
      An[e] = (_Float16)sp_neg;
      As[e] = (_Float16)sg;
    }

    // B tile (32x16) from LDS: same K order as A (lanes 0-15 K={0..7,16..23}, lanes 16-31 K={8..15,24..31}).
    const int4 s0 = *reinterpret_cast<const int4*>(sseg + p0 + kbase);
    const int4 s1 = *reinterpret_cast<const int4*>(sseg + p0 + kbase + 4);
    const int4 s2 = *reinterpret_cast<const int4*>(sseg + p0 + kbase + 16);
    const int4 s3 = *reinterpret_cast<const int4*>(sseg + p0 + kbase + 20);
    int sv[16] = {s0.x, s0.y, s0.z, s0.w, s1.x, s1.y, s1.z, s1.w,
                  s2.x, s2.y, s2.z, s2.w, s3.x, s3.y, s3.z, s3.w};
    v16h B0, B1;
#pragma unroll
    for (int e = 0; e < 16; ++e) {
      int s = sv[e];
      B0[e] = (s == n0 + 1) ? (_Float16)1.0f : (_Float16)0.0f;
      B1[e] = (ones1 || (s == n1 + 1)) ? (_Float16)1.0f : (_Float16)0.0f;
    }

    aP0 = __builtin_amdgcn_wmma_f32_16x16x32_f16(false, Ap, false, B0, (short)0, aP0, false, false);
    aP1 = __builtin_amdgcn_wmma_f32_16x16x32_f16(false, Ap, false, B1, (short)0, aP1, false, false);
    aN0 = __builtin_amdgcn_wmma_f32_16x16x32_f16(false, An, false, B0, (short)0, aN0, false, false);
    aN1 = __builtin_amdgcn_wmma_f32_16x16x32_f16(false, An, false, B1, (short)0, aN1, false, false);
    aI0 = __builtin_amdgcn_wmma_f32_16x16x32_f16(false, As, false, B0, (short)0, aI0, false, false);
    aI1 = __builtin_amdgcn_wmma_f32_16x16x32_f16(false, As, false, B1, (short)0, aI1, false, false);
  }

  __syncthreads();   // all waves done reading sm.seg before reusing LDS as sm.red

  // Dump per-wave partial accumulators (D layout: vgpr r, lane).
#pragma unroll
  for (int r = 0; r < 8; ++r) {
    sm.red[0][wv][r * 32 + lane] = aP0[r];
    sm.red[1][wv][r * 32 + lane] = aP1[r];
    sm.red[2][wv][r * 32 + lane] = aN0[r];
    sm.red[3][wv][r * 32 + lane] = aN1[r];
    sm.red[4][wv][r * 32 + lane] = aI0[r];
    sm.red[5][wv][r * 32 + lane] = aI1[r];
  }
  __syncthreads();

  // Epilogue: threads 0..255 own one (r, lane) cell of the 16x16 D tiles.
  if (t < 256) {
    const int r  = t >> 5;
    const int l2 = t & 31;
    const int tl = (l2 & 16) + 4;          // logical col 20 = tile1 phys col 4
    float P0 = 0, P1 = 0, N0 = 0, N1 = 0, I0 = 0, I1 = 0, negT = 0, sigT = 0;
#pragma unroll
    for (int w = 0; w < WAVES; ++w) {       // fixed order -> deterministic
      P0 += sm.red[0][w][r * 32 + l2];
      P1 += sm.red[1][w][r * 32 + l2];
      N0 += sm.red[2][w][r * 32 + l2];
      N1 += sm.red[3][w][r * 32 + l2];
      I0 += sm.red[4][w][r * 32 + l2];
      I1 += sm.red[5][w][r * 32 + l2];
      negT += sm.red[3][w][r * 32 + tl];    // sum_p softplus(x) for this row
      sigT += sm.red[5][w][r * 32 + tl];    // sum_p sigmoid(x)  for this row
    }
    const int c0 = l2 & 15;                 // tile0 column
    const int c1 = 16 + c0;                 // tile1 column
    const int q  = q0 + (l2 >> 4) * 8 + r;  // D row -> query index
    if (q < Qn) {
      const float gs0 = gsum[b * 32 + c0];
      const float invP = 1.0f / (float)P;
      size_t base = ((size_t)lb * Qn + q) * Nn;
      bce [base + c0] = (P0 + negT - N0) * invP;
      dice[base + c0] = 1.0f - (2.0f * I0 + SMOOTH) / (sigT + gs0 + SMOOTH);
      if (c1 < Nn) {
        const float gs1 = gsum[b * 32 + c1];
        bce [base + c1] = (P1 + negT - N1) * invP;
        dice[base + c1] = 1.0f - (2.0f * I1 + SMOOTH) / (sigT + gs1 + SMOOTH);
      }
    }
  }
}

// ---------------------------------------------------------------------------
// K2: per (l,b,q) row: logsumexp of cls logits (saved for CE) and fused
// final cost matrix = min(5*bce + 5*dice - 2*prob[label], 1e30).
// ---------------------------------------------------------------------------
__global__ __launch_bounds__(32) void k_cls(
    const float* __restrict__ cls, const int* __restrict__ labels,
    const float* __restrict__ bce, const float* __restrict__ dice,
    float* __restrict__ cost, float* __restrict__ lse_ws)
{
  const int row  = blockIdx.x;               // (l*B+b)*Q + q
  const int b    = (row / Qn) % Bn;
  const int lane = threadIdx.x;
  const float* rp = cls + (size_t)row * Cc;

  float mx = -3.0e38f;
  for (int c = lane; c < Cc; c += 32) mx = fmaxf(mx, rp[c]);
#pragma unroll
  for (int off = 16; off >= 1; off >>= 1) mx = fmaxf(mx, __shfl_xor(mx, off, 32));
  float s = 0.0f;
  for (int c = lane; c < Cc; c += 32)
    s += __builtin_amdgcn_exp2f((rp[c] - mx) * LOG2E);
#pragma unroll
  for (int off = 16; off >= 1; off >>= 1) s += __shfl_xor(s, off, 32);
  float lse = mx + __builtin_amdgcn_logf(s) * LN2;
  if (lane == 0) lse_ws[row] = lse;

  if (lane < Nn) {
    int   lab = labels[b * Nn + lane];
    float pr  = __builtin_amdgcn_exp2f((rp[lab] - lse) * LOG2E);
    float cv  = MASK_SCALE * bce [(size_t)row * Nn + lane]
              + DICE_SCALE * dice[(size_t)row * Nn + lane]
              - CE_SCALE * pr;
    cost[(size_t)row * Nn + lane] = fminf(cv, INFC);
  }
}

// ---------------------------------------------------------------------------
// K3: greedy matcher, one block per (l,b). Exact flat-argmin tie-break.
// ---------------------------------------------------------------------------
__global__ __launch_bounds__(128) void k_match(
    const float* __restrict__ cost, int* __restrict__ assign)
{
  __shared__ float c[Qn * Nn];
  __shared__ float bv[128];
  __shared__ int   bi[128];
  const int lb = blockIdx.x;
  const int t  = threadIdx.x;
  for (int i = t; i < Qn * Nn; i += 128) c[i] = cost[(size_t)lb * Qn * Nn + i];
  for (int q = t; q < Qn; q += 128) assign[lb * Qn + q] = -1;
  __syncthreads();
  for (int it = 0; it < Nn; ++it) {
    float best = 3.0e38f; int bidx = Qn * Nn;
    for (int i = t; i < Qn * Nn; i += 128) {
      float v = c[i];
      if (v < best) { best = v; bidx = i; }
    }
    bv[t] = best; bi[t] = bidx;
    __syncthreads();
    if (t == 0) {
      float bb = bv[0]; int ii = bi[0];
      for (int k = 1; k < 128; ++k)
        if (bv[k] < bb || (bv[k] == bb && bi[k] < ii)) { bb = bv[k]; ii = bi[k]; }
      int q = ii / Nn, n = ii % Nn;
      assign[lb * Qn + q] = n;
      for (int j = 0; j < Nn; ++j) c[q * Nn + j] = INFC;
      for (int j = 0; j < Qn; ++j) c[j * Nn + n] = INFC;
    }
    __syncthreads();
  }
}

// ---------------------------------------------------------------------------
// K4: gather matched losses + CE (matched -> GT class, unmatched -> bg=0),
// reduce to the [3, L] output.
// ---------------------------------------------------------------------------
__global__ __launch_bounds__(256) void k_final(
    const float* __restrict__ cls, const int* __restrict__ labels,
    const float* __restrict__ bce, const float* __restrict__ dice,
    const float* __restrict__ lse_ws, const int* __restrict__ assign,
    float* __restrict__ out)
{
  __shared__ float sb[256], sd[256], sc[256];
  const int l = blockIdx.x;
  const int t = threadIdx.x;
  float aB = 0.0f, aD = 0.0f, aC = 0.0f;
  for (int i = t; i < Bn * Qn; i += 256) {
    int b = i / Qn, q = i % Qn;
    int row = (l * Bn + b) * Qn + q;
    int a = assign[row];
    int tgt = 0;
    if (a >= 0) {
      aB += bce [(size_t)row * Nn + a];
      aD += dice[(size_t)row * Nn + a];
      tgt = labels[b * Nn + a];
    }
    aC += lse_ws[row] - cls[(size_t)row * Cc + tgt];   // -log_softmax[tgt]
  }
  sb[t] = aB; sd[t] = aD; sc[t] = aC;
  __syncthreads();
  for (int off = 128; off >= 1; off >>= 1) {
    if (t < off) { sb[t] += sb[t + off]; sd[t] += sd[t + off]; sc[t] += sc[t + off]; }
    __syncthreads();
  }
  if (t == 0) {
    const float num_masks = (float)(Bn * Nn);
    out[0 * L + l] = CE_SCALE   * (sc[0] / (float)(Bn * Qn));
    out[1 * L + l] = MASK_SCALE * (sb[0] / num_masks);
    out[2 * L + l] = DICE_SCALE * (sd[0] / num_masks);
  }
}

// ---------------------------------------------------------------------------
extern "C" void kernel_launch(void* const* d_in, const int* in_sizes, int n_in,
                              void* d_out, int out_size, void* d_ws, size_t ws_size,
                              hipStream_t stream)
{
  const float* mask_logits = (const float*)d_in[0];   // [L,B,Q,H,W] f32
  const float* cls_logits  = (const float*)d_in[1];   // [L,B,Q,C]   f32
  const int*   pan_seg     = (const int*)  d_in[2];   // [B,H,W]     i32
  const int*   labels      = (const int*)  d_in[3];   // [B,N]       i32
  float* out = (float*)d_out;                         // [3,L] f32

  float* ws    = (float*)d_ws;
  float* gsum  = ws;                                  // B*32
  float* bce   = ws + 64;                             // L*B*Q*N
  float* dice  = bce  + (size_t)L * Bn * Qn * Nn;     // L*B*Q*N
  float* cost  = dice + (size_t)L * Bn * Qn * Nn;     // L*B*Q*N
  float* lse   = cost + (size_t)L * Bn * Qn * Nn;     // L*B*Q
  int*   assig = (int*)(lse + (size_t)L * Bn * Qn);   // L*B*Q

  hipLaunchKernelGGL(k_gsum,     dim3(Bn),           dim3(256), 0, stream, pan_seg, gsum);
  hipLaunchKernelGGL(k_pairwise, dim3(L * Bn * QT),  dim3(BLK), 0, stream,
                     mask_logits, pan_seg, gsum, bce, dice);
  hipLaunchKernelGGL(k_cls,      dim3(L * Bn * Qn),  dim3(32),  0, stream,
                     cls_logits, labels, bce, dice, cost, lse);
  hipLaunchKernelGGL(k_match,    dim3(L * Bn),       dim3(128), 0, stream, cost, assig);
  hipLaunchKernelGGL(k_final,    dim3(L),            dim3(256), 0, stream,
                     cls_logits, labels, bce, dice, lse, assig, out);
}
